// ProbabilisticTransformer_21165598835355
// MI455X (gfx1250) — matrix operand order repaired and checked by reading, not compiled, not run on hardware
//
// ProbabilisticTransformer on MI455X (gfx1250, CDNA5, wave32).
// Compute-bound on GEMMs (~530 GFLOP vs ~1GB traffic): everything matrix-shaped
// runs on v_wmma_f32_16x16x32_bf16 (bf16 inputs, fp32 accumulate).
// GEMM global->LDS staging uses the CDNA5 async-to-LDS path (ASYNCcnt) when the
// toolchain exposes the builtin; otherwise a batched sync copy.
#include <hip/hip_runtime.h>
#include <hip/hip_bf16.h>
#include <math.h>

#define BS_   2
#define SEQ_  1024
#define TOK   (BS_*SEQ_)      // 2048 tokens
#define DM    1024
#define D2_   2048
#define FF    4096
#define NH    16
#define DK    64
#define NB    2
#define VOC   32000

typedef __attribute__((ext_vector_type(16))) __bf16 v16bf;
typedef __attribute__((ext_vector_type(8)))  __bf16 v8bf;
typedef __attribute__((ext_vector_type(8)))  float  v8f;
typedef int v4i __attribute__((vector_size(16)));   // matches builtin param type

#define AS1 __attribute__((address_space(1)))
#define AS3 __attribute__((address_space(3)))

#if defined(__AMDGCN__) && __has_builtin(__builtin_amdgcn_global_load_async_to_lds_b128)
#define USE_ASYNC_LDS 1
#else
#define USE_ASYNC_LDS 0
#endif

static __device__ inline void wait_async0() {
#if defined(__AMDGCN__) && __has_builtin(__builtin_amdgcn_s_wait_asynccnt)
  __builtin_amdgcn_s_wait_asynccnt(0);
#else
  asm volatile("s_wait_asynccnt 0x0" ::: "memory");
#endif
}

union V16U { v16bf v; v8bf h[2]; };

static __device__ inline v16bf join16(v8bf lo, v8bf hi) {
  V16U u; u.h[0] = lo; u.h[1] = hi; return u.v;
}
static __device__ inline v8f zero8() {
  v8f z;
#pragma unroll
  for (int i = 0; i < 8; ++i) z[i] = 0.f;
  return z;
}
static __device__ inline float softplus_f(float x) {
  return fmaxf(x, 0.f) + log1pf(__expf(-fabsf(x)));
}

// ---------------------------------------------------------------------------
// Transpose-convert fp32 [K][N] -> bf16 [N][K]  (all dims multiples of 32)
// ---------------------------------------------------------------------------
__global__ __launch_bounds__(256) void cvtT_kernel(const float* __restrict__ src,
                                                   __bf16* __restrict__ dst,
                                                   int K, int N) {
  __shared__ float t[32][33];
  const int bx = blockIdx.x * 32;   // N
  const int by = blockIdx.y * 32;   // K
  const int tx = threadIdx.x & 31;
  const int ty = threadIdx.x >> 5;  // 0..7
#pragma unroll
  for (int r = ty; r < 32; r += 8)
    t[r][tx] = src[(size_t)(by + r) * N + bx + tx];
  __syncthreads();
#pragma unroll
  for (int r = ty; r < 32; r += 8)
    dst[(size_t)(bx + r) * K + by + tx] = (__bf16)t[tx][r];
}

// ---------------------------------------------------------------------------
// Embedding gather + "x = 2*emb[id] + PE(pos)" (faithful to reference), bf16 out
// ---------------------------------------------------------------------------
__global__ void embed_kernel(const int* __restrict__ ids, const float* __restrict__ emb,
                             __bf16* __restrict__ xbf) {
  size_t i = (size_t)blockIdx.x * blockDim.x + threadIdx.x;
  if (i >= (size_t)TOK * D2_) return;
  int t = (int)(i >> 11);        // / 2048
  int d = (int)(i & (D2_ - 1));
  int pos = t & (SEQ_ - 1);
  float e = emb[(size_t)ids[t] * D2_ + d];
  int j2 = d & ~1;               // even channel index
  float freq = __expf(-(float)j2 * (9.210340371976184f / 2048.f)); // ln(1e4)/d_model
  float ang = (float)pos * freq;
  float pe = (d & 1) ? __cosf(ang) : __sinf(ang);
  xbf[i] = (__bf16)(2.f * e + pe);
}

// ---------------------------------------------------------------------------
// LayerNorm over D=1024 (fp32 in, bf16 out), one block per token row
// ---------------------------------------------------------------------------
__global__ __launch_bounds__(256) void ln_kernel(const float* __restrict__ x,
                                                 const float* __restrict__ w,
                                                 const float* __restrict__ b,
                                                 __bf16* __restrict__ out) {
  __shared__ float red[2][256];
  const int row = blockIdx.x;
  const float* xr = x + (size_t)row * DM;
  float v[4], s = 0.f, s2 = 0.f;
#pragma unroll
  for (int k = 0; k < 4; ++k) {
    v[k] = xr[threadIdx.x + k * 256];
    s += v[k]; s2 += v[k] * v[k];
  }
  red[0][threadIdx.x] = s; red[1][threadIdx.x] = s2;
  __syncthreads();
  for (int st = 128; st > 0; st >>= 1) {
    if (threadIdx.x < st) {
      red[0][threadIdx.x] += red[0][threadIdx.x + st];
      red[1][threadIdx.x] += red[1][threadIdx.x + st];
    }
    __syncthreads();
  }
  float mean = red[0][0] * (1.f / DM);
  float var  = red[1][0] * (1.f / DM) - mean * mean;
  float rs = rsqrtf(var + 1e-5f);
#pragma unroll
  for (int k = 0; k < 4; ++k) {
    int c = threadIdx.x + k * 256;
    out[(size_t)row * DM + c] = (__bf16)((v[k] - mean) * rs * w[c] + b[c]);
  }
}

__global__ void softplus_bf_kernel(const float* __restrict__ in, __bf16* __restrict__ out, int n) {
  int i = blockIdx.x * blockDim.x + threadIdx.x;
  if (i < n) out[i] = (__bf16)softplus_f(in[i]);
}

__global__ void cvt_bf_kernel(const float* __restrict__ in, __bf16* __restrict__ out, int n) {
  int i = blockIdx.x * blockDim.x + threadIdx.x;
  if (i < n) out[i] = (__bf16)in[i];
}

// ---------------------------------------------------------------------------
// V pre-transpose: qkv[t][2048 + h*64 + d] -> vt[(b*H+h)*64 + d][seq]
// (gives contiguous-K B fragments for the PV WMMA)
// ---------------------------------------------------------------------------
__global__ void vtrans_kernel(const __bf16* __restrict__ qm, const __bf16* __restrict__ qs,
                              __bf16* __restrict__ vtm, __bf16* __restrict__ vts) {
  int i = blockIdx.x * blockDim.x + threadIdx.x;
  if (i >= BS_ * NH * DK * SEQ_) return;
  int col = i & (SEQ_ - 1);
  int d   = (i >> 10) & (DK - 1);
  int h   = (i >> 16) & (NH - 1);
  int b   = i >> 20;
  size_t src = ((size_t)(b * SEQ_ + col)) * (3 * DM) + 2 * DM + h * DK + d;
  vtm[i] = qm[src];
  vts[i] = qs[src];
}

// ---------------------------------------------------------------------------
// Tiled bf16 WMMA GEMM: C[M][N] = act(A[M][K] @ Bt[N][K]^T + bias) (+residual)
// block = 256 thr = 8 waves; tile 128x128; wave sub-tile 64x32; KT = 64.
// flags: bits0-1 act (0 none,1 softplus,2 relu,3 softplus+1e-6), bit2 residual
// ---------------------------------------------------------------------------
__global__ __launch_bounds__(256) void gemm_bf16_kernel(
    const __bf16* __restrict__ A, const __bf16* __restrict__ Bt,
    const float* __restrict__ bias, float* __restrict__ Cf, __bf16* __restrict__ Cbf,
    int M, int N, int K, int flags)
{
  __shared__ __bf16 sA[128 * 72];   // 128 rows (M) x 64 (K), +8 pad
  __shared__ __bf16 sB[128 * 72];   // 128 rows (N) x 64 (K), +8 pad
  const int tid = threadIdx.x;
  const int wave = tid >> 5, lane = tid & 31;
  const int ml = lane & 15, gl = lane >> 4;
  const int bm = blockIdx.y * 128;
  const int bn = blockIdx.x * 128;
  const int wr = (wave >> 2) * 64;       // wave M base (0,64)
  const int wc = (wave & 3) * 32;        // wave N base (0..96)
  const int lr = tid >> 1;               // staging row 0..127
  const int lc = (tid & 1) * 32;         // staging col 0/32

  v8f acc[4][2];
#pragma unroll
  for (int i = 0; i < 4; ++i)
#pragma unroll
    for (int j = 0; j < 2; ++j) acc[i][j] = zero8();

  const __bf16* ga = A  + (size_t)(bm + lr) * K + lc;
  const __bf16* gb = Bt + (size_t)(bn + lr) * K + lc;

  for (int kk = 0; kk < K; kk += 64) {
    __syncthreads();
#if USE_ASYNC_LDS
    // CDNA5 async DMA global->LDS: no VGPR round trip, tracked by ASYNCcnt.
#pragma unroll
    for (int i = 0; i < 4; ++i) {
      __builtin_amdgcn_global_load_async_to_lds_b128(
          (AS1 v4i*)(AS1 void*)(void*)&ga[kk + i * 8],
          (AS3 v4i*)(AS3 void*)(void*)&sA[lr * 72 + lc + i * 8], 0, 0);
      __builtin_amdgcn_global_load_async_to_lds_b128(
          (AS1 v4i*)(AS1 void*)(void*)&gb[kk + i * 8],
          (AS3 v4i*)(AS3 void*)(void*)&sB[lr * 72 + lc + i * 8], 0, 0);
    }
#else
    // Batched sync copy: issue all 8 global loads before any LDS store so the
    // memory pipe keeps them in flight (avoids per-chunk s_wait_loadcnt 0).
    v8bf ra[4], rb[4];
#pragma unroll
    for (int i = 0; i < 4; ++i) {
      ra[i] = *(const v8bf*)&ga[kk + i * 8];
      rb[i] = *(const v8bf*)&gb[kk + i * 8];
    }
#pragma unroll
    for (int i = 0; i < 4; ++i) {
      *(v8bf*)&sA[lr * 72 + lc + i * 8] = ra[i];
      *(v8bf*)&sB[lr * 72 + lc + i * 8] = rb[i];
    }
#endif
    if (kk + 64 < K) {                       // global_prefetch_b8 next K-slab
      __builtin_prefetch(&ga[kk + 64], 0, 1);
      __builtin_prefetch(&gb[kk + 64], 0, 1);
    }
#if USE_ASYNC_LDS
    wait_async0();
#endif
    __syncthreads();
#pragma unroll
    for (int k0 = 0; k0 < 64; k0 += 32) {
      v16bf af[4], bw[2];
#pragma unroll
      for (int i = 0; i < 4; ++i) {          // A frag: K = e + g*8 (+8 for e>=8)
        const __bf16* p = &sA[(wr + i * 16 + ml) * 72 + k0 + gl * 8];
        af[i] = join16(*(const v8bf*)p, *(const v8bf*)(p + 16));
      }
#pragma unroll
      for (int j = 0; j < 2; ++j) {          // B frag: K = e + g*8 (+8 for e>=8)
        const __bf16* p = &sB[(wc + j * 16 + ml) * 72 + k0 + gl * 8];
        bw[j] = join16(*(const v8bf*)p, *(const v8bf*)(p + 16));
      }
#pragma unroll
      for (int i = 0; i < 4; ++i)
#pragma unroll
        for (int j = 0; j < 2; ++j)
          acc[i][j] = __builtin_amdgcn_wmma_f32_16x16x32_bf16(
              false, af[i], false, bw[j], (short)0, acc[i][j], false, false);
    }
  }

  const int act = flags & 3;
  const bool resid = (flags & 4) != 0;
#pragma unroll
  for (int i = 0; i < 4; ++i) {
#pragma unroll
    for (int j = 0; j < 2; ++j) {
      const int col = bn + wc + j * 16 + ml;
      const float bb = bias ? bias[col] : 0.f;
#pragma unroll
      for (int r = 0; r < 8; ++r) {          // C/D layout: M = r + g*8
        const int row = bm + wr + i * 16 + gl * 8 + r;
        float v = acc[i][j][r] + bb;
        if (act == 1)      v = softplus_f(v);
        else if (act == 2) v = fmaxf(v, 0.f);
        else if (act == 3) v = softplus_f(v) + 1e-6f;
        size_t idx = (size_t)row * N + col;
        if (Cf) { if (resid) v += Cf[idx]; Cf[idx] = v; }
        if (Cbf) Cbf[idx] = (__bf16)v;
      }
    }
  }
}

// ---------------------------------------------------------------------------
// Fused distance-kernel attention (flash-style, no max-rescale needed since
// sim = exp(-W) <= 1).  Grid: (SEQ/64, BS*NH), 128 threads = 4 waves, each
// wave owns 16 query rows.  Per 32-key block: 8 WMMAs for S (mu+sg dots
// chained into one accumulator), LDS C->A relayout of P, 8 WMMAs for PV.
// ---------------------------------------------------------------------------
__global__ __launch_bounds__(128) void attn_kernel(
    const __bf16* __restrict__ qkvm, const __bf16* __restrict__ qkvs,
    const __bf16* __restrict__ vtm,  const __bf16* __restrict__ vts,
    __bf16* __restrict__ om, __bf16* __restrict__ os)
{
  __shared__ __bf16 sP[4][16 * 32];   // per-wave P staging (1KB each)
  const int wave = threadIdx.x >> 5;
  const int lane = threadIdx.x & 31;
  const int ml = lane & 15;
  const int gl = lane >> 4;
  const int bh = blockIdx.y;
  const int b = bh >> 4, h = bh & 15;
  const int q0 = blockIdx.x * 64 + wave * 16;
  const int LD = 3 * DM;
  const __bf16* Qm = qkvm + (size_t)(b * SEQ_) * LD + h * DK;
  const __bf16* Qs = qkvs + (size_t)(b * SEQ_) * LD + h * DK;
  const __bf16* Km = Qm + DM;
  const __bf16* Ks = Qs + DM;
  __bf16* spw = sP[wave];

  // Q fragments (A-matrix, 16x32 per dk-half) + query norms |Qm|^2+|Qs|^2
  v16bf aQm[2], aQs[2];
  float qpart = 0.f;
#pragma unroll
  for (int kk = 0; kk < 2; ++kk) {
    const __bf16* pm = &Qm[(size_t)(q0 + ml) * LD + kk * 32 + gl * 8];
    const __bf16* ps = &Qs[(size_t)(q0 + ml) * LD + kk * 32 + gl * 8];
    aQm[kk] = join16(*(const v8bf*)pm, *(const v8bf*)(pm + 16));
    aQs[kk] = join16(*(const v8bf*)ps, *(const v8bf*)(ps + 16));
#pragma unroll
    for (int e = 0; e < 16; ++e) {
      float a = (float)aQm[kk][e]; qpart += a * a;
      float c = (float)aQs[kk][e]; qpart += c * c;
    }
  }
  float qn = qpart + __shfl_xor(qpart, 16);   // full row norm on both halves

  v8f oam[4], oas[4];
#pragma unroll
  for (int i = 0; i < 4; ++i) { oam[i] = zero8(); oas[i] = zero8(); }
  float rsum[8];
#pragma unroll
  for (int r = 0; r < 8; ++r) rsum[r] = 0.f;

  for (int j0 = 0; j0 <= q0 + 15; j0 += 32) {   // causal: only keys <= q0+15
    v8f sacc[2]; float kn[2];
#pragma unroll
    for (int nt = 0; nt < 2; ++nt) {
      sacc[nt] = zero8();
      float kp = 0.f;
#pragma unroll
      for (int kk = 0; kk < 2; ++kk) {
        const __bf16* pm = &Km[(size_t)(j0 + nt * 16 + ml) * LD + kk * 32 + gl * 8];
        const __bf16* ps = &Ks[(size_t)(j0 + nt * 16 + ml) * LD + kk * 32 + gl * 8];
        v16bf bm = join16(*(const v8bf*)pm, *(const v8bf*)(pm + 16));
        v16bf bs = join16(*(const v8bf*)ps, *(const v8bf*)(ps + 16));
#pragma unroll
        for (int e = 0; e < 16; ++e) {
          float a = (float)bm[e]; kp += a * a;
          float c = (float)bs[e]; kp += c * c;
        }
        sacc[nt] = __builtin_amdgcn_wmma_f32_16x16x32_bf16(
            false, aQm[kk], false, bm, (short)0, sacc[nt], false, false);
        sacc[nt] = __builtin_amdgcn_wmma_f32_16x16x32_bf16(
            false, aQs[kk], false, bs, (short)0, sacc[nt], false, false);
      }
      kn[nt] = kp + __shfl_xor(kp, 16);
    }
    // d2 -> sim -> p, stage 16x32 P tile in LDS (relayout C-frag -> A-frag)
#pragma unroll
    for (int nt = 0; nt < 2; ++nt) {
      int jcol = j0 + nt * 16 + ml;
#pragma unroll
      for (int r = 0; r < 8; ++r) {
        int qrow = gl * 8 + r;
        float qn_r = __shfl(qn, qrow);             // row-norm broadcast
        float d2 = qn_r + kn[nt] - 2.f * sacc[nt][r];
        float w = sqrtf(fmaxf(d2, 1e-12f));
        float sim = __expf(-w);
        float p = (jcol <= q0 + qrow) ? __expf(sim) : 0.f;
        rsum[r] += p;
        spw[qrow * 32 + nt * 16 + ml] = (__bf16)p;
      }
    }
    // wave-private LDS handshake: DS ops are in-order per wave, but make the
    // cross-lane visibility explicit with the CDNA5 split counter.
    asm volatile("s_wait_dscnt 0x0" ::: "memory");
    const __bf16* pp = &spw[ml * 32 + gl * 8];
    v16bf pf = join16(*(const v8bf*)pp, *(const v8bf*)(pp + 16));
#pragma unroll
    for (int nv = 0; nv < 4; ++nv) {
      const __bf16* vm = &vtm[((size_t)bh * DK + nv * 16 + ml) * SEQ_ + j0 + gl * 8];
      const __bf16* vs = &vts[((size_t)bh * DK + nv * 16 + ml) * SEQ_ + j0 + gl * 8];
      v16bf bvm = join16(*(const v8bf*)vm, *(const v8bf*)(vm + 16));
      v16bf bvs = join16(*(const v8bf*)vs, *(const v8bf*)(vs + 16));
      oam[nv] = __builtin_amdgcn_wmma_f32_16x16x32_bf16(
          false, pf, false, bvm, (short)0, oam[nv], false, false);
      oas[nv] = __builtin_amdgcn_wmma_f32_16x16x32_bf16(
          false, pf, false, bvs, (short)0, oas[nv], false, false);
    }
  }

  // softmax denominators: reduce across the 16 lanes of each half-wave
#pragma unroll
  for (int r = 0; r < 8; ++r) {
    float s = rsum[r];
    s += __shfl_xor(s, 1); s += __shfl_xor(s, 2);
    s += __shfl_xor(s, 4); s += __shfl_xor(s, 8);
    rsum[r] = 1.f / s;
  }
#pragma unroll
  for (int nv = 0; nv < 4; ++nv) {
    int col = h * DK + nv * 16 + ml;
#pragma unroll
    for (int r = 0; r < 8; ++r) {
      size_t tok = (size_t)(b * SEQ_ + q0 + gl * 8 + r);
      om[tok * DM + col] = (__bf16)(oam[nv][r] * rsum[r]);
      os[tok * DM + col] = (__bf16)(oas[nv][r] * rsum[r]);
    }
  }
}

// ---------------------------------------------------------------------------
// Host orchestration
// ---------------------------------------------------------------------------
extern "C" void kernel_launch(void* const* d_in, const int* in_sizes, int n_in,
                              void* d_out, int out_size, void* d_ws, size_t ws_size,
                              hipStream_t stream) {
  (void)in_sizes; (void)n_in; (void)out_size; (void)ws_size;
  const int*   ids       = (const int*)d_in[0];
  const float* emb       = (const float*)d_in[1];
  const float* pemb_mu_w = (const float*)d_in[2];
  const float* pemb_mu_b = (const float*)d_in[3];
  const float* pemb_sg_w = (const float*)d_in[4];
  const float* pemb_sg_b = (const float*)d_in[5];
  const float* qkv_mu_w  = (const float*)d_in[6];
  const float* qkv_mu_b  = (const float*)d_in[7];
  const float* qkv_sg_w  = (const float*)d_in[8];
  const float* qkv_sg_b  = (const float*)d_in[9];
  const float* out_w     = (const float*)d_in[10];
  const float* out_b     = (const float*)d_in[11];
  const float* ln1_w     = (const float*)d_in[12];
  const float* ln1_b     = (const float*)d_in[13];
  const float* ln2_w     = (const float*)d_in[14];
  const float* ln2_b     = (const float*)d_in[15];
  const float* ffm_w1    = (const float*)d_in[16];
  const float* ffm_b1    = (const float*)d_in[17];
  const float* ffm_w2    = (const float*)d_in[18];
  const float* ffm_b2    = (const float*)d_in[19];
  const float* ffs_w1    = (const float*)d_in[20];
  const float* ffs_b1    = (const float*)d_in[21];
  const float* ffs_w2    = (const float*)d_in[22];
  const float* ffs_b2    = (const float*)d_in[23];
  const float* out_mu_w  = (const float*)d_in[24];
  const float* out_mu_b  = (const float*)d_in[25];
  const float* out_sg_w  = (const float*)d_in[26];

  char* wsb = (char*)d_ws;
  size_t off = 0;
  auto alloc = [&](size_t elems, size_t esz) -> void* {
    off = (off + 255) & ~(size_t)255;
    void* p = wsb + off;
    off += elems * esz;
    return p;
  };
  // bf16 transposed weight caches ([N][K] so GEMM B-fragments are contiguous)
  __bf16* wPembMu = (__bf16*)alloc((size_t)DM * D2_, 2);
  __bf16* wPembSg = (__bf16*)alloc((size_t)DM * D2_, 2);
  __bf16* wQkvMu  = (__bf16*)alloc((size_t)NB * 3 * DM * DM, 2);
  __bf16* wQkvSg  = (__bf16*)alloc((size_t)NB * 3 * DM * DM, 2);
  __bf16* wOut    = (__bf16*)alloc((size_t)NB * DM * DM, 2);
  __bf16* wFfm1   = (__bf16*)alloc((size_t)NB * FF * DM, 2);
  __bf16* wFfm2   = (__bf16*)alloc((size_t)NB * DM * FF, 2);
  __bf16* wFfs1   = (__bf16*)alloc((size_t)NB * FF * DM, 2);
  __bf16* wFfs2   = (__bf16*)alloc((size_t)NB * DM * FF, 2);
  __bf16* wOutMu  = (__bf16*)alloc((size_t)VOC * DM, 2);
  __bf16* wOutSg  = (__bf16*)alloc((size_t)VOC * DM, 2);
  // activations
  __bf16* xbf    = (__bf16*)alloc((size_t)TOK * D2_, 2);
  float*  muF    = (float*) alloc((size_t)TOK * DM, 4);
  float*  sgF    = (float*) alloc((size_t)TOK * DM, 4);
  __bf16* nmuBf  = (__bf16*)alloc((size_t)TOK * DM, 2);
  __bf16* ssigBf = (__bf16*)alloc((size_t)TOK * DM, 2);
  __bf16* qkvmBf = (__bf16*)alloc((size_t)TOK * 3 * DM, 2);
  __bf16* qkvsBf = (__bf16*)alloc((size_t)TOK * 3 * DM, 2);
  __bf16* vtm    = (__bf16*)alloc((size_t)TOK * DM, 2);
  __bf16* vts    = (__bf16*)alloc((size_t)TOK * DM, 2);
  __bf16* omBf   = (__bf16*)alloc((size_t)TOK * DM, 2);
  __bf16* osBf   = (__bf16*)alloc((size_t)TOK * DM, 2);
  __bf16* hidBf  = (__bf16*)alloc((size_t)TOK * FF, 2);
  __bf16* muBf   = (__bf16*)alloc((size_t)TOK * DM, 2);
  __bf16* sgBf   = (__bf16*)alloc((size_t)TOK * DM, 2);

  auto cvtT = [&](const float* s, __bf16* d, int K, int N) {
    cvtT_kernel<<<dim3(N / 32, K / 32), 256, 0, stream>>>(s, d, K, N);
  };
  auto gemm = [&](const __bf16* A, const __bf16* Bt, const float* bias,
                  float* Cf, __bf16* Cbf, int M, int N, int K, int flags) {
    gemm_bf16_kernel<<<dim3(N / 128, M / 128), 256, 0, stream>>>(
        A, Bt, bias, Cf, Cbf, M, N, K, flags);
  };

  // weight conversion (deterministic, every call)
  cvtT(pemb_mu_w, wPembMu, D2_, DM);
  cvtT(pemb_sg_w, wPembSg, D2_, DM);
  for (int l = 0; l < NB; ++l) {
    cvtT(qkv_mu_w + (size_t)l * DM * 3 * DM, wQkvMu + (size_t)l * 3 * DM * DM, DM, 3 * DM);
    cvtT(qkv_sg_w + (size_t)l * DM * 3 * DM, wQkvSg + (size_t)l * 3 * DM * DM, DM, 3 * DM);
    cvtT(out_w    + (size_t)l * DM * DM,     wOut   + (size_t)l * DM * DM,     DM, DM);
    cvtT(ffm_w1   + (size_t)l * DM * FF,     wFfm1  + (size_t)l * FF * DM,     DM, FF);
    cvtT(ffm_w2   + (size_t)l * FF * DM,     wFfm2  + (size_t)l * DM * FF,     FF, DM);
    cvtT(ffs_w1   + (size_t)l * DM * FF,     wFfs1  + (size_t)l * FF * DM,     DM, FF);
    cvtT(ffs_w2   + (size_t)l * FF * DM,     wFfs2  + (size_t)l * DM * FF,     FF, DM);
  }
  cvtT(out_mu_w, wOutMu, DM, VOC);
  cvtT(out_sg_w, wOutSg, DM, VOC);

  embed_kernel<<<(TOK * D2_) / 256, 256, 0, stream>>>(ids, emb, xbf);

  gemm(xbf, wPembMu, pemb_mu_b, muF, nullptr, TOK, DM, D2_, 0);   // mu
  gemm(xbf, wPembSg, pemb_sg_b, sgF, nullptr, TOK, DM, D2_, 1);   // softplus(sigma)

  for (int l = 0; l < NB; ++l) {
    ln_kernel<<<TOK, 256, 0, stream>>>(muF, ln1_w + l * DM, ln1_b + l * DM, nmuBf);
    softplus_bf_kernel<<<(TOK * DM) / 256, 256, 0, stream>>>(sgF, ssigBf, TOK * DM);
    gemm(nmuBf, wQkvMu + (size_t)l * 3 * DM * DM, qkv_mu_b + l * 3 * DM,
         nullptr, qkvmBf, TOK, 3 * DM, DM, 0);
    gemm(ssigBf, wQkvSg + (size_t)l * 3 * DM * DM, qkv_sg_b + l * 3 * DM,
         nullptr, qkvsBf, TOK, 3 * DM, DM, 0);
    vtrans_kernel<<<(TOK * DM) / 256, 256, 0, stream>>>(qkvmBf, qkvsBf, vtm, vts);
    attn_kernel<<<dim3(SEQ_ / 64, BS_ * NH), 128, 0, stream>>>(
        qkvmBf, qkvsBf, vtm, vts, omBf, osBf);
    gemm(omBf, wOut + (size_t)l * DM * DM, out_b + l * DM, muF, nullptr, TOK, DM, DM, 4);
    gemm(osBf, wOut + (size_t)l * DM * DM, out_b + l * DM, sgF, nullptr, TOK, DM, DM, 4);
    ln_kernel<<<TOK, 256, 0, stream>>>(muF, ln2_w + l * DM, ln2_b + l * DM, nmuBf);
    softplus_bf_kernel<<<(TOK * DM) / 256, 256, 0, stream>>>(sgF, ssigBf, TOK * DM);
    gemm(nmuBf, wFfm1 + (size_t)l * FF * DM, ffm_b1 + l * FF, nullptr, hidBf, TOK, FF, DM, 2);
    gemm(hidBf, wFfm2 + (size_t)l * DM * FF, ffm_b2 + l * DM, muF, nullptr, TOK, DM, FF, 4);
    gemm(ssigBf, wFfs1 + (size_t)l * FF * DM, ffs_b1 + l * FF, nullptr, hidBf, TOK, FF, DM, 2);
    gemm(hidBf, wFfs2 + (size_t)l * DM * FF, ffs_b2 + l * DM, sgF, nullptr, TOK, DM, FF, 4);
  }

  cvt_bf_kernel<<<(TOK * DM) / 256, 256, 0, stream>>>(muF, muBf, TOK * DM);
  cvt_bf_kernel<<<(TOK * DM) / 256, 256, 0, stream>>>(sgF, sgBf, TOK * DM);

  float* logits   = (float*)d_out;
  float* logitsSg = logits + (size_t)TOK * VOC;
  gemm(muBf, wOutMu, out_mu_b, logits,   nullptr, TOK, VOC, DM, 0);  // mu head
  gemm(sgBf, wOutSg, nullptr,  logitsSg, nullptr, TOK, VOC, DM, 3);  // softplus+1e-6
}
